// AttentionGate_33938831573711
// MI455X (gfx1250) — matrix operand order repaired and checked by reading, not compiled, not run on hardware
//
#include <hip/hip_runtime.h>

// ---------------------------------------------------------------------------
// AttentionGate for MI455X (gfx1250, wave32, WMMA bf16 16x16x32 + TDM)
//   g,x: [2,320,32,64,64] f32 ; Wg,Wx: [160,320] ; Wpsi: [1,160]
//   out = x * sigmoid(Wpsi . (elu(IN(Wg g + bg)) + elu(IN(Wx x + bx))) + bpsi)
// 3-kernel pipeline: WMMA stats pass -> tiny reduce -> fused WMMA apply pass.
// Tiles are staged global->LDS with tensor_load_to_lds (TDM), double-buffered
// against TENSORcnt; A-fragments are software-pipelined against the WMMAs.
// ---------------------------------------------------------------------------

typedef unsigned short u16;
typedef __attribute__((ext_vector_type(16))) __bf16        v16bf;
typedef __attribute__((ext_vector_type(8)))  float         v8f;
typedef __attribute__((ext_vector_type(4)))  float         v4f;
typedef __attribute__((ext_vector_type(4)))  unsigned int  v4u;
typedef __attribute__((ext_vector_type(8)))  unsigned int  v8u;

constexpr int  CIN   = 320;     // K (input channels)
constexpr int  FINT  = 160;     // M (inter channels)
constexpr long SP    = 131072;  // spatial per batch = 32*64*64
constexpr int  NB    = 256;     // column-blocks per (n,which)
constexpr int  COLSB = 512;     // columns per block (4 tiles of 128)
constexpr int  TSR   = 132;     // padded LDS row stride in floats (128 + 4 via TDM pad)
constexpr int  TILEB = 32 * TSR * 4;   // 16896 bytes per K-chunk tile

union BF16Frag { unsigned u[8]; v16bf v; };

// One v_perm_b32: D = { b[31:16], a[31:16] }  (f32 -> bf16 truncate, packed)
__device__ __forceinline__ unsigned pack_bf16x2(float a, float b) {
    return __builtin_amdgcn_perm(__builtin_bit_cast(unsigned, b),
                                 __builtin_bit_cast(unsigned, a), 0x07060302u);
}

__device__ __forceinline__ v8f wmma_bf16(v16bf a, v16bf b, v8f c) {
    return __builtin_amdgcn_wmma_f32_16x16x32_bf16(false, a, false, b, (short)0, c, false, false);
}

__device__ __forceinline__ void wait_tensorcnt0() {
#if __has_builtin(__builtin_amdgcn_s_wait_tensorcnt)
    __builtin_amdgcn_s_wait_tensorcnt((short)0);
#else
    asm volatile("s_wait_tensorcnt 0x0" ::: "memory");
#endif
}

// ---------------------------------------------------------------------------
// TDM descriptor (ISA §8.3/8.4): one 32(K-rows) x 128(cols) f32 tile into LDS,
// row-major with a 4-dword pad every 128 dwords (LDS row stride 132).
// Group1 is loop-invariant -> built once and kept in SGPRs.
// ---------------------------------------------------------------------------
__device__ __forceinline__ v8u tdm_make_group1() {
    v8u g1;
    g1[0] = (2u << 16)                                 // data_size = 4B
          | (1u << 20)                                 // pad_enable
          | (6u << 22)                                 // pad_interval: 128 dwords
          | (3u << 25);                                // pad_amount:   4 dwords
    g1[1] = ((unsigned)(SP & 0xFFFFu)) << 16;          // tensor_dim0[15:0]
    g1[2] = (unsigned)(SP >> 16)                       // tensor_dim0[31:16]
          | ((unsigned)CIN << 16);                     // tensor_dim1[15:0]
    g1[3] = (128u << 16);                              // tensor_dim1 hi | tile_dim0=128
    g1[4] = 32u;                                       // tile_dim1=32 (tile_dim2=0)
    g1[5] = (unsigned)SP;                              // tensor_dim0_stride[31:0]
    g1[6] = 0u;
    g1[7] = 0u;
    return g1;
}

__device__ __forceinline__ void tdm_issue(unsigned lds_off, const float* gsrc, v8u g1) {
    unsigned long long ga = (unsigned long long)(const void*)gsrc;
    v4u g0;
    g0[0] = 1u;                                        // count=1, user D#
    g0[1] = lds_off;                                   // lds_addr (bytes)
    g0[2] = (unsigned)ga;                              // global_addr[31:0]
    g0[3] = (unsigned)((ga >> 32) & 0x01FFFFFFu)       // global_addr[56:32]
          | (2u << 30);                                // type=2 ("image")
    asm volatile("tensor_load_to_lds %0, %1" :: "s"(g0), "s"(g1) : "memory");
}

// K index inside a 16x32 bf16 A-fragment for (lane, half i) per ISA 7.12.2
__device__ __forceinline__ int a_k_of(int lane, int i) {
    int hi = lane >> 4;
    int v  = i >> 1, lo = i & 1;
    int base = (v < 4) ? (2 * v) : (16 + 2 * (v - 4));
    return base + lo + hi * 8;
}

// Pre-swizzle W[160x320] f32 -> LDS bf16 in exact A-fragment order.
__device__ __forceinline__ void preload_wfrag(u16* dst, const float* __restrict__ W) {
    for (int idx = threadIdx.x; idx < FINT * CIN; idx += 256) {
        int i    = idx & 15;
        int lane = (idx >> 4) & 31;
        int fc   = idx >> 9;            // m*10 + kc
        int kc   = fc % 10;
        int m    = fc / 10;
        int o    = m * 16 + (lane & 15);
        int k    = kc * 32 + a_k_of(lane, i);
        unsigned bits = __builtin_bit_cast(unsigned, W[o * CIN + k]);
        dst[idx] = (u16)(bits >> 16);
    }
}

// B fragment (32K x 16N bf16) from the row-major padded tile:
// col = lane%16, k in ISA table order (8-k group per lane half); stride 132 is conflict-free in k.
__device__ __forceinline__ v16bf load_bfrag(const float* tile, int colL, int koff) {
    BF16Frag f;
#pragma unroll
    for (int q = 0; q < 8; ++q) {
        float a = tile[((koff >> 1) + 2 * q + ((q >= 4) ? 8 : 0) + 0) * TSR + colL];
        float b = tile[((koff >> 1) + 2 * q + ((q >= 4) ? 8 : 0) + 1) * TSR + colL];
        f.u[q] = pack_bf16x2(a, b);
    }
    return f.v;
}

__device__ __forceinline__ v16bf load_afrag(const u16* src) {
    const uint4* s = (const uint4*)src;
    uint4 a = s[0], b = s[1];
    BF16Frag f;
    f.u[0] = a.x; f.u[1] = a.y; f.u[2] = a.z; f.u[3] = a.w;
    f.u[4] = b.x; f.u[5] = b.y; f.u[6] = b.z; f.u[7] = b.w;
    return f.v;
}

// ---------------------------------------------------------------------------
// Core: y[0:160, 16-col tile] = W * t[:, cols]  with TDM double-buffered K loop
// and a 2-stage A-fragment rotation (ds_load(m+1) issues before wmma(m)).
// ---------------------------------------------------------------------------
__device__ __forceinline__ void gemm_tile(
    v8f acc[10], const float* __restrict__ src, long pt,
    const u16* __restrict__ wfrag,
    const float* tile0, const float* tile1, unsigned off0, unsigned off1,
    int lane, int colL, int koff, bool issuer, v8u g1)
{
    __syncthreads();                                   // tiles free for reuse
    if (issuer) tdm_issue(off0, src + pt, g1);         // kc = 0
#pragma unroll 2
    for (int kc = 0; kc < 10; ++kc) {
        if (issuer) wait_tensorcnt0();                 // chunk kc landed in LDS
        __syncthreads();                               // publish to all waves
        if (issuer && kc < 9)                          // overlap DMA kc+1 w/ math
            tdm_issue((kc & 1) ? off0 : off1,
                      src + (long)((kc + 1) * 32) * SP + pt, g1);
        const float* cur = (kc & 1) ? tile1 : tile0;
        v16bf bf = load_bfrag(cur, colL, koff);
        v16bf af = load_afrag(wfrag + ((0 * 10 + kc) << 9) + (lane << 4));
#pragma unroll
        for (int m = 0; m < 10; ++m) {
            v16bf afn = af;
            if (m < 9)
                afn = load_afrag(wfrag + (((m + 1) * 10 + kc) << 9) + (lane << 4));
            acc[m] = wmma_bf16(af, bf, acc[m]);
            af = afn;
        }
    }
}

// ---------------------------------------------------------------------------
// Kernel 1: per-block partial (sum, sumsq) of y = W*t + b over 512 columns.
// grid = (NB, n=2, which=2), block = 256 (8 waves)
// ---------------------------------------------------------------------------
__global__ __launch_bounds__(256) void ag_stats_kernel(
    const float* __restrict__ g, const float* __restrict__ x,
    const float* __restrict__ Wg, const float* __restrict__ bg,
    const float* __restrict__ Wx, const float* __restrict__ bx,
    float* __restrict__ partials)
{
    extern __shared__ char smem[];
    u16*   wfrag  = (u16*)smem;                           // 100 KB
    float* tile0  = (float*)(smem + 102400);              // 16896 B
    float* tile1  = (float*)(smem + 102400 + TILEB);      // 16896 B
    float* statsS = (float*)(smem + 102400 + 2 * TILEB);  // 320 f32
    float* biasS  = statsS + 320;                         // 160 f32
    const unsigned off0 = 102400u, off1 = 102400u + TILEB;

    const int which = blockIdx.z;
    const int n     = blockIdx.y;
    const float* src0 = (which ? x : g) + (long)n * CIN * SP;
    const float* W    = which ? Wx : Wg;
    const float* bias = which ? bx : bg;

    preload_wfrag(wfrag, W);
    for (int i = threadIdx.x; i < 320; i += 256) statsS[i] = 0.0f;
    for (int i = threadIdx.x; i < 160; i += 256) biasS[i] = bias[i];
    __syncthreads();

    const int  lane   = threadIdx.x & 31;
    const int  wave   = threadIdx.x >> 5;
    const int  colL   = wave * 16 + (lane & 15);
    const int  koff   = (lane & 16) ? 16 : 0;
    const int  hi     = lane >> 4;
    const bool issuer = (threadIdx.x < 32);
    const long p0     = (long)blockIdx.x * COLSB;
    const v8u  g1     = tdm_make_group1();

    for (int t = 0; t < 4; ++t) {
        const long pt = p0 + (long)t * 128;
        v8f acc[10] = {};
        gemm_tile(acc, src0, pt, wfrag, tile0, tile1, off0, off1,
                  lane, colL, koff, issuer, g1);
        // column reduction: shfl butterfly over 16 column-lanes, then 1-lane
        // LDS atomic per (o, component) -> no 16-way atomic conflicts.
#pragma unroll
        for (int m = 0; m < 10; ++m) {
#pragma unroll
            for (int r = 0; r < 8; ++r) {
                int   o = m * 16 + r + 8 * hi;
                float v = acc[m][r] + biasS[o];
                float s = v, q = v * v;
#pragma unroll
                for (int msk = 1; msk < 16; msk <<= 1) {
                    s += __shfl_xor(s, msk, 32);
                    q += __shfl_xor(q, msk, 32);
                }
                if ((lane & 15) == 0) {
                    atomicAdd(&statsS[o * 2 + 0], s);     // ds_add_f32
                    atomicAdd(&statsS[o * 2 + 1], q);
                }
            }
        }
    }
    __syncthreads();
    float* dst = partials + ((long)((which * 2 + n) * NB + blockIdx.x)) * 320;
    for (int i = threadIdx.x; i < 320; i += 256) dst[i] = statsS[i];
}

// ---------------------------------------------------------------------------
// Kernel 2: reduce NB partials -> (mean, rstd) per (which, n, o). 640 items.
// ---------------------------------------------------------------------------
__global__ void ag_reduce_kernel(const float* __restrict__ partials,
                                 float* __restrict__ stats)
{
    int idx = blockIdx.x * blockDim.x + threadIdx.x;
    if (idx >= 2 * 2 * FINT) return;
    int o   = idx % FINT;
    int rem = idx / FINT;               // which*2 + n
    float s = 0.0f, ss = 0.0f;
    for (int b = 0; b < NB; ++b) {
        const float* p = partials + ((long)(rem * NB + b)) * 320 + o * 2;
        s  += p[0];
        ss += p[1];
    }
    const float inv = 1.0f / (float)SP;
    float mean = s * inv;
    float var  = ss * inv - mean * mean;
    float rstd = rsqrtf(var + 1e-5f);
    stats[rem * 320 + o * 2 + 0] = mean;
    stats[rem * 320 + o * 2 + 1] = rstd;
}

// ---------------------------------------------------------------------------
// Kernel 3: fused recompute-GEMM + norm + ELU + psi-dot + sigmoid + x*psi.
// Two sequential phases (G then X) keep live accumulators at 80 VGPRs.
// grid = (NB, n=2), block = 256
// ---------------------------------------------------------------------------
__global__ __launch_bounds__(256) void ag_fused_kernel(
    const float* __restrict__ g, const float* __restrict__ x,
    const float* __restrict__ Wg, const float* __restrict__ bg,
    const float* __restrict__ Wx, const float* __restrict__ bx,
    const float* __restrict__ Wpsi, const float* __restrict__ bpsi,
    const float* __restrict__ stats, float* __restrict__ out)
{
    extern __shared__ char smem[];
    u16*   wfragG = (u16*)smem;                               // 100 KB
    u16*   wfragX = (u16*)(smem + 102400);                    // 100 KB
    float* tile0  = (float*)(smem + 204800);                  // 16896 B
    float* tile1  = (float*)(smem + 204800 + TILEB);          // 16896 B
    float* sG     = (float*)(smem + 204800 + 2 * TILEB);      // 5 x 160 f32
    float* tG     = sG + 160;
    float* sX     = tG + 160;
    float* tX     = sX + 160;
    float* wpsiS  = tX + 160;
    float* psiS   = wpsiS + 160;                              // 128 f32
    const unsigned off0 = 204800u, off1 = 204800u + TILEB;

    const int n = blockIdx.y;

    preload_wfrag(wfragG, Wg);
    preload_wfrag(wfragX, Wx);
    if (threadIdx.x < FINT) {
        int o = threadIdx.x;
        float mg = stats[(0 * 2 + n) * 320 + o * 2 + 0];
        float rg = stats[(0 * 2 + n) * 320 + o * 2 + 1];
        float mx = stats[(1 * 2 + n) * 320 + o * 2 + 0];
        float rx = stats[(1 * 2 + n) * 320 + o * 2 + 1];
        sG[o] = rg;  tG[o] = (bg[o] - mg) * rg;   // IN(acc+b) = acc*rstd + (b-mean)*rstd
        sX[o] = rx;  tX[o] = (bx[o] - mx) * rx;
        wpsiS[o] = Wpsi[o];
    }
    __syncthreads();

    const float* gsrc = g + (long)n * CIN * SP;
    const float* xsrc = x + (long)n * CIN * SP;
    float*       outp = out + (long)n * CIN * SP;
    const float  bpsiV = bpsi[0];

    const int  lane   = threadIdx.x & 31;
    const int  wave   = threadIdx.x >> 5;
    const int  colL   = wave * 16 + (lane & 15);
    const int  koff   = (lane & 16) ? 16 : 0;
    const int  hi     = lane >> 4;
    const bool issuer = (threadIdx.x < 32);
    const long p0     = (long)blockIdx.x * COLSB;
    const v8u  g1     = tdm_make_group1();

    for (int t = 0; t < 4; ++t) {
        const long pt = p0 + (long)t * 128;
        float dot = 0.0f;

        {   // ---- G phase ----
            v8f acc[10] = {};
            gemm_tile(acc, gsrc, pt, wfragG, tile0, tile1, off0, off1,
                      lane, colL, koff, issuer, g1);
#pragma unroll
            for (int m = 0; m < 10; ++m)
#pragma unroll
                for (int r = 0; r < 8; ++r) {
                    int   o = m * 16 + r + 8 * hi;
                    float y = acc[m][r] * sG[o] + tG[o];
                    float e = (y > 0.0f) ? y : (__expf(y) - 1.0f);
                    dot += e * wpsiS[o];
                }
        }
        {   // ---- X phase ----
            v8f acc[10] = {};
            gemm_tile(acc, xsrc, pt, wfragX, tile0, tile1, off0, off1,
                      lane, colL, koff, issuer, g1);
#pragma unroll
            for (int m = 0; m < 10; ++m)
#pragma unroll
                for (int r = 0; r < 8; ++r) {
                    int   o = m * 16 + r + 8 * hi;
                    float y = acc[m][r] * sX[o] + tX[o];
                    float e = (y > 0.0f) ? y : (__expf(y) - 1.0f);
                    dot += e * wpsiS[o];
                }
        }

        dot += __shfl_xor(dot, 16, 32);                 // combine lane halves (wave32)
        float psi = 1.0f / (1.0f + __expf(-(dot + bpsiV)));
        if (lane < 16) psiS[wave * 16 + lane] = psi;
        __syncthreads();

        // out[:, pt..pt+127] = x * psi   (coalesced, non-temporal stores)
#pragma unroll 8
        for (int i = 0; i < 40; ++i) {
            int lin = threadIdx.x + i * 256;            // 0..10239 = 320*32
            int c   = lin >> 5;
            int c4  = (lin & 31) << 2;
            v4f xv = *(const v4f*)(xsrc + (long)c * SP + pt + c4);
            v4f pv = *(const v4f*)(psiS + c4);
            __builtin_nontemporal_store(xv * pv, (v4f*)(outp + (long)c * SP + pt + c4));
        }
        __syncthreads();
    }
}

// ---------------------------------------------------------------------------
extern "C" void kernel_launch(void* const* d_in, const int* in_sizes, int n_in,
                              void* d_out, int out_size, void* d_ws, size_t ws_size,
                              hipStream_t stream) {
    const float* g    = (const float*)d_in[0];
    const float* x    = (const float*)d_in[1];
    const float* Wg   = (const float*)d_in[2];
    const float* bg   = (const float*)d_in[3];
    const float* Wx   = (const float*)d_in[4];
    const float* bx   = (const float*)d_in[5];
    const float* Wpsi = (const float*)d_in[6];
    const float* bpsi = (const float*)d_in[7];
    float* out = (float*)d_out;

    float* partials = (float*)d_ws;                        // 2*2*NB*320 f32 = 1.31 MB
    float* stats    = partials + (long)2 * 2 * NB * 320;   // 1280 f32

    const size_t shmemStats = 102400 + 2 * TILEB + (320 + 160) * 4;
    const size_t shmemFused = 204800 + 2 * TILEB + (160 * 5 + 128) * 4;

    ag_stats_kernel<<<dim3(NB, 2, 2), 256, shmemStats, stream>>>(
        g, x, Wg, bg, Wx, bx, partials);
    ag_reduce_kernel<<<3, 256, 0, stream>>>(partials, stats);
    ag_fused_kernel<<<dim3(NB, 2), 256, shmemFused, stream>>>(
        g, x, Wg, bg, Wx, bx, Wpsi, bpsi, stats, out);
}
